// GRUDecoder_14877766714158
// MI455X (gfx1250) — matrix operand order repaired and checked by reading, not compiled, not run on hardware
//
#include <hip/hip_runtime.h>
#include <hip/hip_bf16.h>

// ---------------------------------------------------------------------------
// GRU decoder for MI455X (gfx1250, wave32, WMMA).
//   S0: split x, enc to bf16 hi/lo      (one-time, removes A-side cvt VALU)
//   K0: GI    = x   @ Wih^T + bih       (WMMA split-bf16 GEMM, 512x3072x768)
//   P1: WaT   = Wa^T                    (tiled transpose)
//   P2: EB0   = enc . ba                (matvec)
//   K3: EncWa = enc @ Wa                (WMMA GEMM, 1024x1024x1024)
//   K4: WcEnc = enc @ Wc[:, :1024]^T    (WMMA GEMM, 1024x1024x1024, ldb=2048)
//   K5: persistent 64-WG scan, 3 grid barriers/step, emits Hgru hi/lo bf16
//   K6: out   = Hgru @ Wp^T + bp        (WMMA split-bf16 GEMM, 512x28996x1024)
// ---------------------------------------------------------------------------

typedef __attribute__((ext_vector_type(16))) __bf16 bf16x16;
typedef __attribute__((ext_vector_type(8)))  __bf16 bf16x8;
typedef __attribute__((ext_vector_type(8)))  float  floatx8;

#define GRUD_T     512
#define GRUD_IN    768
#define GRUD_H     1024
#define GRUD_L     128
#define GRUD_VOCAB 28996
#define SCAN_WGS   64

// ---------------------------- GEMM (split-bf16 WMMA) ------------------------

__device__ __forceinline__ void load8f(const float* __restrict__ p, float* d) {
  const float4 a = *reinterpret_cast<const float4*>(p);
  const float4 b = *reinterpret_cast<const float4*>(p + 4);
  d[0] = a.x; d[1] = a.y; d[2] = a.z; d[3] = a.w;
  d[4] = b.x; d[5] = b.y; d[6] = b.z; d[7] = b.w;
}

__device__ __forceinline__ void split16(const float* f, bf16x16& hi, bf16x16& lo) {
#pragma unroll
  for (int e = 0; e < 16; ++e) {
    __bf16 h = (__bf16)f[e];
    hi[e] = h;
    lo[e] = (__bf16)(f[e] - (float)h);
  }
}

// A fragment (16x32 bf16, ISA 7.12.2): p = row_base + kk + kbA.
// elems 0..7  -> K = kbA + e        (p[0..7],  16 B)
// elems 8..15 -> K = 16 + kbA + e-8 (p[16..23], 16 B)
__device__ __forceinline__ bf16x16 load_afrag(const __bf16* __restrict__ p) {
  bf16x8 q0 = *reinterpret_cast<const bf16x8*>(p);
  bf16x8 q1 = *reinterpret_cast<const bf16x8*>(p + 16);
  return __builtin_shufflevector(q0, q1, 0, 1, 2, 3, 4, 5, 6, 7,
                                 8, 9, 10, 11, 12, 13, 14, 15);
}

// C[M,N] = A[M,K] @ Bw[N,:K of ldb]^T (+ bias[N]).
// A pre-split to bf16 hi/lo (Ah/Al, leading dim K).  K % 32 == 0, M % 512 == 0.
// Block = 256 threads (8 waves); tile 512(M) x 64(N): 4 M-chunks in
// accumulators so each B element is loaded+split exactly once per wave;
// N-subtiles processed in 2 groups of 2 to keep the live set under 256 VGPRs.
// grid = (ceil(N/64), M/512).
__global__ __launch_bounds__(256)
void gemm_bf16x3_kernel(const __bf16* __restrict__ Ah,
                        const __bf16* __restrict__ Al,
                        const float* __restrict__ Bw,
                        const float* __restrict__ bias,
                        float* __restrict__ C,
                        int M, int N, int K, int ldb) {
  const int lane   = threadIdx.x & 31;
  const int wv     = threadIdx.x >> 5;        // 0..7
  const int hrow   = lane & 15;               // M (A) / N (B,C) index
  const int kbA    = (lane >> 4) << 3;        // A frag K-base: 0 or 8
  const int kbB    = (lane >> 4) << 3;        // B frag K-base: 0 or 8
  const int nBase  = blockIdx.x * 64;
  const int mBase0 = blockIdx.y * 512;

  floatx8 acc[4][4];
#pragma unroll
  for (int mc = 0; mc < 4; ++mc)
#pragma unroll
    for (int ns = 0; ns < 4; ++ns)
#pragma unroll
      for (int r = 0; r < 8; ++r) acc[mc][ns][r] = 0.0f;

  size_t aoff[4];
#pragma unroll
  for (int mc = 0; mc < 4; ++mc)
    aoff[mc] = (size_t)(mBase0 + mc * 128 + wv * 16 + hrow) * K + kbA;

  bool bval[4]; int cols[4]; const float* bptr[4];
#pragma unroll
  for (int ns = 0; ns < 4; ++ns) {
    int c    = nBase + ns * 16 + hrow;
    bval[ns] = (c < N);
    cols[ns] = c;
    bptr[ns] = Bw + (size_t)(bval[ns] ? c : 0) * ldb + kbB;
  }

  for (int kk = 0; kk < K; kk += 32) {
#pragma unroll
    for (int nsg = 0; nsg < 2; ++nsg) {
      // ---- B fragments (32x16 bf16: lane = col, lane-half picks K range;
      //      elems 0..7 -> K = kbB + e, elems 8..15 -> K = 16 + kbB + e-8)
      bf16x16 bh[2], bl[2];
#pragma unroll
      for (int g2 = 0; g2 < 2; ++g2) {
        const int ns = nsg * 2 + g2;
        float bfr[16];
        load8f(bptr[ns] + kk,     bfr);
        load8f(bptr[ns] + kk + 16, bfr + 8);
        if (!bval[ns]) {
#pragma unroll
          for (int e = 0; e < 16; ++e) bfr[e] = 0.0f;
        }
        split16(bfr, bh[g2], bl[g2]);
        if (kk + 32 < K) __builtin_prefetch(bptr[ns] + kk + 32, 0, 0);
      }

#pragma unroll
      for (int mc = 0; mc < 4; ++mc) {
        const bf16x16 ah = load_afrag(Ah + aoff[mc] + kk);
        const bf16x16 al = load_afrag(Al + aoff[mc] + kk);
        // split-precision fp32: A*B ~= ah*bh + ah*bl + al*bh (fp32 accum)
#pragma unroll
        for (int g2 = 0; g2 < 2; ++g2) {
          const int ns = nsg * 2 + g2;
          acc[mc][ns] = __builtin_amdgcn_wmma_f32_16x16x32_bf16(
              false, ah, false, bh[g2], (short)0, acc[mc][ns], false, false);
          acc[mc][ns] = __builtin_amdgcn_wmma_f32_16x16x32_bf16(
              false, ah, false, bl[g2], (short)0, acc[mc][ns], false, false);
          acc[mc][ns] = __builtin_amdgcn_wmma_f32_16x16x32_bf16(
              false, al, false, bh[g2], (short)0, acc[mc][ns], false, false);
        }
      }
    }
  }

  // C/D layout: lane holds N=hrow; VGPR r -> M = r (+8 for upper lane half)
  const int rbump = (lane >> 4) << 3;
#pragma unroll
  for (int ns = 0; ns < 4; ++ns) {
    if (!bval[ns]) continue;
    const float bv = bias ? bias[cols[ns]] : 0.0f;
#pragma unroll
    for (int mc = 0; mc < 4; ++mc) {
#pragma unroll
      for (int r = 0; r < 8; ++r) {
        const int row = mBase0 + mc * 128 + wv * 16 + r + rbump;
        C[(size_t)row * N + cols[ns]] = acc[mc][ns][r] + bv;
      }
    }
  }
}

// ------------------------------ prep kernels --------------------------------

// fp32 -> bf16 hi/lo split (one element per thread).
__global__ __launch_bounds__(256)
void split_f32_kernel(const float* __restrict__ src, __bf16* __restrict__ hi,
                      __bf16* __restrict__ lo, int n) {
  const int i = blockIdx.x * 256 + threadIdx.x;
  if (i < n) {
    const float v = src[i];
    const __bf16 h = (__bf16)v;
    hi[i] = h;
    lo[i] = (__bf16)(v - (float)h);
  }
}

// dst[k][1024] = src[j][1024] transposed (1024x1024), 32x32 LDS tiles.
__global__ __launch_bounds__(256)
void transpose1024_kernel(const float* __restrict__ src, float* __restrict__ dst) {
  __shared__ float t[32][33];
  const int bx = blockIdx.x * 32, by = blockIdx.y * 32;
  const int x = threadIdx.x;       // 0..31
#pragma unroll
  for (int yy = threadIdx.y; yy < 32; yy += 8)
    t[yy][x] = src[(size_t)(by + yy) * 1024 + bx + x];
  __syncthreads();
#pragma unroll
  for (int yy = threadIdx.y; yy < 32; yy += 8)
    dst[(size_t)(bx + yy) * 1024 + by + x] = t[x][yy];
}

__device__ __forceinline__ float red16(float v) {
  v += __shfl_xor(v, 8, 32);
  v += __shfl_xor(v, 4, 32);
  v += __shfl_xor(v, 2, 32);
  v += __shfl_xor(v, 1, 32);
  return v;
}

// EB0[i] = enc_flat[i] . ba   for i in [0,1024)  (enc_flat = [B*L][H])
__global__ __launch_bounds__(256)
void eb0_kernel(const float* __restrict__ enc, const float* __restrict__ ba,
                float* __restrict__ EB0) {
  const int grp = threadIdx.x >> 4, sub = threadIdx.x & 15;
  const int i = blockIdx.x * 16 + grp;
  const float* e = enc + (size_t)i * GRUD_H;
  float s = 0.f;
  for (int k = sub * 64; k < sub * 64 + 64; ++k) s = fmaf(e[k], ba[k], s);
  s = red16(s);
  if (sub == 0) EB0[i] = s;
}

// ------------------------------ sequential scan -----------------------------

__device__ __forceinline__ float sigm(float x) {
  return 1.0f / (1.0f + __expf(-x));
}

__device__ __forceinline__ void gbar(unsigned* bar, unsigned nb, unsigned& rnd) {
  __threadfence();
  __syncthreads();
  if (threadIdx.x == 0) {
    ++rnd;
    atomicAdd(bar, 1u);
    const unsigned target = rnd * nb;
    while (__hip_atomic_load(bar, __ATOMIC_RELAXED, __HIP_MEMORY_SCOPE_AGENT)
           < target) {
      __builtin_amdgcn_s_sleep(1);
    }
  }
  __syncthreads();
  __threadfence();
}

__global__ __launch_bounds__(256)
void gru_scan_kernel(const float* __restrict__ GI,    // [T][3H] (bih folded)
                     const float* __restrict__ h0,    // [H]
                     const float* __restrict__ Whh,   // [3H][H]
                     const float* __restrict__ bhh,   // [3H]
                     const float* __restrict__ EncWa, // [B*L][H]
                     const float* __restrict__ EB0,   // [B*L]
                     const float* __restrict__ WcEnc, // [B*L][H]
                     const float* __restrict__ Wc,    // [H][2H] (use 2nd half)
                     const float* __restrict__ bc,    // [H]
                     float* __restrict__ h_buf,
                     float* __restrict__ hg_buf,
                     float* __restrict__ sc_buf,
                     __bf16* __restrict__ Hgru_hi,    // [T][H]
                     __bf16* __restrict__ Hgru_lo,    // [T][H]
                     unsigned* __restrict__ bar) {
  const int tid = threadIdx.x;
  const int g   = blockIdx.x;                 // 0..63
  const unsigned NB = gridDim.x;              // 64
  unsigned rnd = 0;
  const int grp = tid >> 4;                   // 0..15
  const int sub = tid & 15;                   // 0..15 (one wave half)
  const int j   = g * 16 + grp;               // owned hidden index

  __shared__ float sscore[GRUD_L];
  __shared__ float red[256];

  { int gt = g * 256 + tid; if (gt < GRUD_H) h_buf[gt] = h0[gt]; }
  gbar(bar, NB, rnd);

  for (int t = 0; t < GRUD_T; ++t) {
    const int bid = t >> 6;                   // W = 64 words per sample

    // ---- phase A: gh = Whh@h, gates, h_gru -------------------------------
    {
      const float* wr = Whh + (size_t)j * GRUD_H;
      const float* wz = Whh + (size_t)(j + GRUD_H) * GRUD_H;
      const float* wn = Whh + (size_t)(j + 2 * GRUD_H) * GRUD_H;
      float sr = 0.f, sz = 0.f, sn = 0.f;
      for (int k = sub * 64; k < sub * 64 + 64; ++k) {
        const float hv = h_buf[k];
        sr = fmaf(wr[k], hv, sr);
        sz = fmaf(wz[k], hv, sz);
        sn = fmaf(wn[k], hv, sn);
      }
      sr = red16(sr); sz = red16(sz); sn = red16(sn);
      if (sub == 0) {
        const float* gi = GI + (size_t)t * (3 * GRUD_H);
        const float r = sigm(gi[j] + sr + bhh[j]);
        const float z = sigm(gi[GRUD_H + j] + sz + bhh[GRUD_H + j]);
        const float n = tanhf(gi[2 * GRUD_H + j] +
                              r * (sn + bhh[2 * GRUD_H + j]));
        const float hg = (1.0f - z) * n + z * h_buf[j];
        hg_buf[j] = hg;
        const __bf16 hh = (__bf16)hg;
        Hgru_hi[(size_t)t * GRUD_H + j] = hh;
        Hgru_lo[(size_t)t * GRUD_H + j] = (__bf16)(hg - (float)hh);
      }
    }
    gbar(bar, NB, rnd);

    // ---- phase B: scores[l] = EncWa[bid,l].h_gru + EB0[bid,l] ------------
    {
      const int lg = tid >> 7;                // 0..1
      const int ss = tid & 127;
      const int l  = g * 2 + lg;
      const size_t row = (size_t)bid * GRUD_L + l;
      const float* e = EncWa + row * GRUD_H;
      float p = 0.f;
      for (int k = ss * 8; k < ss * 8 + 8; ++k)
        p = fmaf(e[k], hg_buf[k], p);
      red[tid] = p;
      __syncthreads();
      for (int st = 64; st > 0; st >>= 1) {
        if ((tid & 127) < st) red[tid] += red[tid + st];
        __syncthreads();
      }
      if ((tid & 127) == 0) sc_buf[l] = red[tid] + EB0[row];
    }
    gbar(bar, NB, rnd);

    // ---- phase C: softmax; h = tanh(sum_l p_l WcEnc[l] + Wc2@h_gru + bc) -
    {
      if (tid < GRUD_L) sscore[tid] = sc_buf[tid];
      __syncthreads();
      float mx = -3.0e38f;
      for (int l = 0; l < GRUD_L; ++l) mx = fmaxf(mx, sscore[l]);
      float den = 0.f;
      for (int l = 0; l < GRUD_L; ++l) den += __expf(sscore[l] - mx);

      float p1 = 0.f;
      for (int l = sub * 8; l < sub * 8 + 8; ++l)
        p1 = fmaf(__expf(sscore[l] - mx),
                  WcEnc[((size_t)bid * GRUD_L + l) * GRUD_H + j], p1);
      const float* w2 = Wc + (size_t)j * (2 * GRUD_H) + GRUD_H;
      float p2 = 0.f;
      for (int m = sub * 64; m < sub * 64 + 64; ++m)
        p2 = fmaf(w2[m], hg_buf[m], p2);
      p1 = red16(p1);
      p2 = red16(p2);
      if (sub == 0) h_buf[j] = tanhf(p1 / den + p2 + bc[j]);
      __syncthreads();
    }
    gbar(bar, NB, rnd);
  }
}

// ------------------------------- launch ------------------------------------

extern "C" void kernel_launch(void* const* d_in, const int* in_sizes, int n_in,
                              void* d_out, int out_size, void* d_ws,
                              size_t ws_size, hipStream_t stream) {
  (void)in_sizes; (void)n_in; (void)out_size; (void)ws_size;

  const float* x   = (const float*)d_in[0];   // [8,64,768]
  const float* h0  = (const float*)d_in[1];   // [1,1024]
  const float* enc = (const float*)d_in[2];   // [8,128,1024] (flat [1024,1024])
  const float* Wih = (const float*)d_in[3];   // [3072,768]
  const float* Whh = (const float*)d_in[4];   // [3072,1024]
  const float* bih = (const float*)d_in[5];   // [3072]
  const float* bhh = (const float*)d_in[6];   // [3072]
  const float* Wp  = (const float*)d_in[7];   // [28996,1024]
  const float* bp  = (const float*)d_in[8];   // [28996]
  const float* Wa  = (const float*)d_in[9];   // [1024,1024]
  const float* ba  = (const float*)d_in[10];  // [1024]
  const float* Wc  = (const float*)d_in[11];  // [1024,2048]
  const float* bc  = (const float*)d_in[12];  // [1024]
  float* out = (float*)d_out;                 // [512,28996]

  char* ws = (char*)d_ws;
  size_t off = 0;
  auto alloc = [&](size_t bytes) {
    void* p = ws + off;
    off += (bytes + 255) & ~(size_t)255;
    return p;
  };
  unsigned* bar   = (unsigned*)alloc(256);
  float* h_buf    = (float*)alloc(GRUD_H * 4);
  float* hg_buf   = (float*)alloc(GRUD_H * 4);
  float* sc_buf   = (float*)alloc(GRUD_L * 4);
  float* EB0      = (float*)alloc(1024 * 4);
  float* GI       = (float*)alloc((size_t)GRUD_T * 3 * GRUD_H * 4);   // 6 MB
  float* WaT      = (float*)alloc((size_t)1024 * 1024 * 4);           // 4 MB
  float* EncWa    = (float*)alloc((size_t)1024 * 1024 * 4);           // 4 MB
  float* WcEnc    = (float*)alloc((size_t)1024 * 1024 * 4);           // 4 MB
  __bf16* Hgru_hi = (__bf16*)alloc((size_t)GRUD_T * GRUD_H * 2);      // 1 MB
  __bf16* Hgru_lo = (__bf16*)alloc((size_t)GRUD_T * GRUD_H * 2);      // 1 MB
  __bf16* x_hi    = (__bf16*)alloc((size_t)GRUD_T * GRUD_IN * 2);
  __bf16* x_lo    = (__bf16*)alloc((size_t)GRUD_T * GRUD_IN * 2);
  __bf16* enc_hi  = (__bf16*)alloc((size_t)1024 * 1024 * 2);          // 2 MB
  __bf16* enc_lo  = (__bf16*)alloc((size_t)1024 * 1024 * 2);          // 2 MB

  hipMemsetAsync(bar, 0, 256, stream);   // grid-barrier counter (per replay)

  const dim3 blk(256);
  // S0: one-time bf16 hi/lo splits of the A-side operands
  {
    const int nx = GRUD_T * GRUD_IN;
    split_f32_kernel<<<dim3((nx + 255) / 256), blk, 0, stream>>>(x, x_hi, x_lo, nx);
    const int ne = 1024 * 1024;
    split_f32_kernel<<<dim3((ne + 255) / 256), blk, 0, stream>>>(enc, enc_hi, enc_lo, ne);
  }
  // K0: GI = x @ Wih^T + bih                     [512 x 3072 x 768]
  gemm_bf16x3_kernel<<<dim3(3 * GRUD_H / 64, 1), blk, 0, stream>>>(
      x_hi, x_lo, Wih, bih, GI, GRUD_T, 3 * GRUD_H, GRUD_IN, GRUD_IN);
  // P1: WaT = Wa^T
  transpose1024_kernel<<<dim3(32, 32), dim3(32, 8), 0, stream>>>(Wa, WaT);
  // P2: EB0 = enc . ba
  eb0_kernel<<<dim3(64), blk, 0, stream>>>(enc, ba, EB0);
  // K3: EncWa = enc @ WaT^T  (i.e. enc @ Wa)     [1024 x 1024 x 1024]
  gemm_bf16x3_kernel<<<dim3(1024 / 64, 2), blk, 0, stream>>>(
      enc_hi, enc_lo, WaT, nullptr, EncWa, 1024, 1024, GRUD_H, GRUD_H);
  // K4: WcEnc = enc @ Wc[:, :1024]^T             [1024 x 1024 x 1024]
  gemm_bf16x3_kernel<<<dim3(1024 / 64, 2), blk, 0, stream>>>(
      enc_hi, enc_lo, Wc, nullptr, WcEnc, 1024, 1024, GRUD_H, 2 * GRUD_H);
  // K5: sequential scan (persistent, 3 grid barriers per step)
  gru_scan_kernel<<<dim3(SCAN_WGS), blk, 0, stream>>>(
      GI, h0, Whh, bhh, EncWa, EB0, WcEnc, Wc, bc,
      h_buf, hg_buf, sc_buf, Hgru_hi, Hgru_lo, bar);
  // K6: out = Hgru @ Wp^T + bp                   [512 x 28996 x 1024]
  gemm_bf16x3_kernel<<<dim3((GRUD_VOCAB + 63) / 64, 1), blk, 0, stream>>>(
      Hgru_hi, Hgru_lo, Wp, bp, out, GRUD_T, GRUD_VOCAB, GRUD_H, GRUD_H);
}
